// GatedDeltaNet_25666724561458
// MI455X (gfx1250) — matrix operand order repaired and checked by reading, not compiled, not run on hardware
//
#include <hip/hip_runtime.h>

// ---------------- problem constants ----------------
#define BB   2
#define LLEN 2048
#define DD   2048
#define HH   12
#define DKK  1536
#define DVV  3072
#define HKD  128
#define HVD  256
#define CKC  64
#define NCH  32

typedef __bf16 bf16_t;
typedef __attribute__((ext_vector_type(16))) __bf16 bf16x16;
typedef __attribute__((ext_vector_type(8)))  __bf16 bf16x8;
typedef __attribute__((ext_vector_type(8)))  float  v8f;
typedef __attribute__((ext_vector_type(4))) unsigned u32x4;
typedef __attribute__((ext_vector_type(8))) int      i32x8;
typedef __attribute__((ext_vector_type(4))) int      i32x4;

// ---------------- WMMA helper ----------------
__device__ __forceinline__ v8f wmma_bf16(bf16x16 a, bf16x16 b, v8f c) {
  // D = A(16x32) * B(32x16) + C, fp32 accumulate
  return __builtin_amdgcn_wmma_f32_16x16x32_bf16(false, a, false, b, (short)0, c,
                                                 false, false);
}

// ---------------- TDM: 2-D tile load Global -> LDS ----------------
// D# per CDNA5 ISA. data_size=2 (bf16). Optional LDS
// row padding so a 32-elem (64B = 16 DWORD) row lands on a 40-elem (80B) stride:
// pad_interval code 3 (=16 DWORDs), pad_amount code 3 (=4 DWORDs).
__device__ __forceinline__ void tdm_load_tile_2d(unsigned lds_byte_addr,
                                                 const bf16_t* gptr,
                                                 unsigned rows_total, unsigned cols_total,
                                                 unsigned tile_rows, unsigned tile_cols,
                                                 bool pad_en) {
  unsigned long long ga = (unsigned long long)(size_t)gptr;
  unsigned g1w0 = (1u << 16) |                                   // data_size = 2 bytes
                  (pad_en ? ((1u << 20) | (3u << 22) | (3u << 25)) : 0u);
  u32x4 g0 = { 1u,                                               // count=1, user D#
               lds_byte_addr,                                    // lds_addr [63:32]
               (unsigned)(ga & 0xFFFFFFFFu),                     // global_addr [95:64]
               (unsigned)((ga >> 32) & 0x01FFFFFFu) | (2u << 30) }; // [120:96] + type=2
  i32x8 g1 = { (int)g1w0,
               (int)((cols_total & 0xFFFFu) << 16),              // tensor_dim0 [63:48]
               (int)((cols_total >> 16) | ((rows_total & 0xFFFFu) << 16)), // dim0 hi | dim1 lo
               (int)(((rows_total >> 16) & 0xFFFFu) | (tile_cols << 16)),  // dim1 hi | tile_dim0
               (int)tile_rows,                                   // tile_dim1 [143:128]
               (int)cols_total,                                  // dim0_stride lo [191:160]
               (int)(cols_total << 16),                          // stride0 hi | dim1_stride lo
               (int)(cols_total >> 16) };                        // dim1_stride hi
  i32x4 z4 = {0, 0, 0, 0};
#if __clang_major__ >= 23
  i32x8 z8 = {0, 0, 0, 0, 0, 0, 0, 0};
  __builtin_amdgcn_tensor_load_to_lds(g0, g1, z4, z4, z8, 0);
#else
  __builtin_amdgcn_tensor_load_to_lds(g0, g1, z4, z4, 0);
#endif
}

// ---- fragment loaders (layouts per CDNA5 ISA 7.12.2) ----
// A 16x32 bf16: lane m=lane&15; elems 0..7 -> K=(lane>=16?8:0)+e ; 8..15 -> 16+(lane>=16?8:0)+e-8
__device__ __forceinline__ bf16x16 frag_a_lds(const bf16_t* p, int ld, int m0, int k0,
                                              int lane) {
  int m  = m0 + (lane & 15);
  int kb = (lane >> 4) * 8;
  union { bf16x16 v; bf16x8 h[2]; } u;
  u.h[0] = *(const bf16x8*)(p + m * ld + k0 + kb);
  u.h[1] = *(const bf16x8*)(p + m * ld + k0 + 16 + kb);
  return u.v;
}
// B 32x16 from W stored row-major [N,K] (i.e. B = W^T): lane n=lane&15; elems 0..7 -> K=(lane>=16?8:0)+e ; 8..15 -> 16+(lane>=16?8:0)+e-8
__device__ __forceinline__ bf16x16 frag_b_lds(const bf16_t* p, int ld, int n0, int k0,
                                              int lane) {
  int n  = n0 + (lane & 15);
  int kb = (lane >> 4) * 8;
  union { bf16x16 v; bf16x8 h[2]; } u;
  u.h[0] = *(const bf16x8*)(p + n * ld + k0 + kb);
  u.h[1] = *(const bf16x8*)(p + n * ld + k0 + 16 + kb);
  return u.v;
}
// generic gather loaders (arbitrary strides / fp32 sources, scan kernel only)
__device__ __forceinline__ bf16x16 frag_a_bf(const bf16_t* __restrict__ p, int rs, int ks,
                                             int m0, int k0, int lane) {
  int m = m0 + (lane & 15);
  int kh = (lane >> 4) * 8;
  bf16x16 r;
#pragma unroll
  for (int e = 0; e < 16; ++e) {
    int kk = (e < 8) ? (kh + e) : (16 + kh + (e - 8));
    r[e] = p[m * rs + (k0 + kk) * ks];
  }
  return r;
}
__device__ __forceinline__ bf16x16 frag_a_f32(const float* __restrict__ p, int rs, int ks,
                                              int m0, int k0, int lane) {
  int m = m0 + (lane & 15);
  int kh = (lane >> 4) * 8;
  bf16x16 r;
#pragma unroll
  for (int e = 0; e < 16; ++e) {
    int kk = (e < 8) ? (kh + e) : (16 + kh + (e - 8));
    r[e] = (bf16_t)p[m * rs + (k0 + kk) * ks];
  }
  return r;
}
__device__ __forceinline__ bf16x16 frag_b_f32(const float* __restrict__ p, int ks, int ns,
                                              int k0, int n0, int lane) {
  int n = n0 + (lane & 15);
  int kb = (lane >> 4) * 8;
  bf16x16 r;
#pragma unroll
  for (int e = 0; e < 16; ++e) r[e] = (bf16_t)p[(k0 + ((e < 8) ? (kb + e) : (16 + kb + (e - 8)))) * ks + n * ns];
  return r;
}

// ---------------- casts / packing ----------------
__global__ void cast_f32_to_bf16(const float* __restrict__ s, bf16_t* __restrict__ d, long n) {
  long stride = (long)gridDim.x * blockDim.x;
  for (long i = (long)blockIdx.x * blockDim.x + threadIdx.x; i < n; i += stride)
    d[i] = (bf16_t)s[i];
}

// w: [C,C,4] f32 -> d: [4][C][C] bf16 (per-tap row-major matrices)
__global__ void pack_conv_w(const float* __restrict__ w, bf16_t* __restrict__ d, int Cc) {
  long total = (long)Cc * Cc * 4;
  long stride = (long)gridDim.x * blockDim.x;
  for (long i = (long)blockIdx.x * blockDim.x + threadIdx.x; i < total; i += stride) {
    int t = (int)(i & 3);
    long cc = i >> 2;            // co*Cc + ci
    long co = cc / Cc;
    int ci = (int)(cc - co * Cc);
    d[((long)t * Cc + co) * Cc + ci] = (bf16_t)w[i];
  }
}

// ------- generic GEMM: C[M,N] = A[M,K] @ W[N,K]^T ; tiles staged via TDM -------
template <typename OutT>
__global__ __launch_bounds__(256) void gemm_nt(const bf16_t* __restrict__ A,
                                               const bf16_t* __restrict__ W,
                                               OutT* __restrict__ C,
                                               int M, int N, int K) {
  __shared__ bf16_t sA[128 * 40];   // LDS offset 0      (TDM destination)
  __shared__ bf16_t sB[128 * 40];   // LDS offset 10240  (TDM destination)
  const int tid = threadIdx.x, lane = tid & 31, wid = tid >> 5;
  const int m_base = blockIdx.y * 128, n_base = blockIdx.x * 128;
  const int wm = wid & 1, wn = wid >> 1;
  const int rhi = (lane >> 4) * 8, cnl = lane & 15;

  // opaque (never-taken) store so LLVM cannot treat the TDM-written LDS as undef
  if (K < 0) { sA[0] = (bf16_t)0.f; sB[0] = (bf16_t)0.f; }

  v8f acc[4][2] = {};

  for (int k0 = 0; k0 < K; k0 += 32) {
    __syncthreads();                    // previous iteration's reads done
    if (wid == 0) {                     // one wave issues both tensor DMAs
      tdm_load_tile_2d(0u,     A + (long)m_base * K + k0, (unsigned)M, (unsigned)K,
                       128u, 32u, true);
      tdm_load_tile_2d(10240u, W + (long)n_base * K + k0, (unsigned)N, (unsigned)K,
                       128u, 32u, true);
    }
    __builtin_amdgcn_s_wait_tensorcnt(0);
    __syncthreads();                    // tiles visible to all waves
    bf16x16 af[4], bfr[2];
#pragma unroll
    for (int mi = 0; mi < 4; ++mi) af[mi] = frag_a_lds(sA, 40, wm * 64 + mi * 16, 0, lane);
#pragma unroll
    for (int ni = 0; ni < 2; ++ni) bfr[ni] = frag_b_lds(sB, 40, wn * 32 + ni * 16, 0, lane);
#pragma unroll
    for (int mi = 0; mi < 4; ++mi)
#pragma unroll
      for (int ni = 0; ni < 2; ++ni)
        acc[mi][ni] = wmma_bf16(af[mi], bfr[ni], acc[mi][ni]);
  }
#pragma unroll
  for (int mi = 0; mi < 4; ++mi)
#pragma unroll
    for (int ni = 0; ni < 2; ++ni)
#pragma unroll
      for (int r = 0; r < 8; ++r) {
        int row = m_base + wm * 64 + mi * 16 + rhi + r;
        int col = n_base + wn * 32 + ni * 16 + cnl;
        C[(long)row * N + col] = (OutT)acc[mi][ni][r];
      }
}

// ------- causal full-channel conv as 4 accumulated GEMMs with shifted rows -------
// X: bf16 [B*L, C]; Wt: bf16 [4][C][C]; Y: f32 [B*L, C]  (manual staging: needs
// the per-row causal shift / zero-fill, which a plain TDM tile cannot express)
__global__ __launch_bounds__(256) void conv_gemm(const bf16_t* __restrict__ X,
                                                 const bf16_t* __restrict__ Wt,
                                                 float* __restrict__ Y, int Cc) {
  __shared__ bf16_t sA[128 * 40];
  __shared__ bf16_t sB[128 * 40];
  const int tid = threadIdx.x, lane = tid & 31, wid = tid >> 5;
  const int m_base = blockIdx.y * 128, n_base = blockIdx.x * 128;
  const int wm = wid & 1, wn = wid >> 1;
  const int rhi = (lane >> 4) * 8, cnl = lane & 15;
  const int srow = tid >> 1, shalf = (tid & 1) * 16;
  const int gr = m_base + srow;
  const int lpos = gr & (LLEN - 1);   // tiles never cross batch boundary (2048%128==0)

  v8f acc[4][2] = {};
  for (int t = 0; t < 4; ++t) {
    const bf16_t* Wtap = Wt + (long)t * Cc * Cc;
    const int shift = t - 3;
    const bool valid = (lpos + shift) >= 0;    // causal zero pad
    for (int k0 = 0; k0 < Cc; k0 += 32) {
      __syncthreads();
      if (valid) {
        *(int4*)(sA + srow * 40 + shalf) =
            *(const int4*)(X + (long)(gr + shift) * Cc + k0 + shalf);
      } else {
        int4 z = {0, 0, 0, 0};
        *(int4*)(sA + srow * 40 + shalf) = z;
      }
      *(int4*)(sB + srow * 40 + shalf) =
          *(const int4*)(Wtap + (long)(n_base + srow) * Cc + k0 + shalf);
      __syncthreads();
      bf16x16 af[4], bfr[2];
#pragma unroll
      for (int mi = 0; mi < 4; ++mi) af[mi] = frag_a_lds(sA, 40, wm * 64 + mi * 16, 0, lane);
#pragma unroll
      for (int ni = 0; ni < 2; ++ni) bfr[ni] = frag_b_lds(sB, 40, wn * 32 + ni * 16, 0, lane);
#pragma unroll
      for (int mi = 0; mi < 4; ++mi)
#pragma unroll
        for (int ni = 0; ni < 2; ++ni)
          acc[mi][ni] = wmma_bf16(af[mi], bfr[ni], acc[mi][ni]);
    }
  }
#pragma unroll
  for (int mi = 0; mi < 4; ++mi)
#pragma unroll
    for (int ni = 0; ni < 2; ++ni)
#pragma unroll
      for (int r = 0; r < 8; ++r) {
        int row = m_base + wm * 64 + mi * 16 + rhi + r;
        int col = n_base + wn * 32 + ni * 16 + cnl;
        Y[(long)row * Cc + col] = acc[mi][ni][r];
      }
}

// ---------------- mamba gates: gk, beta  [B*L, H] ----------------
__global__ __launch_bounds__(256) void gate_kernel(const float* __restrict__ hs,
                                                   const float* __restrict__ Wgk,
                                                   const float* __restrict__ Wb,
                                                   const float* __restrict__ b_b,
                                                   const float* __restrict__ A_log,
                                                   const float* __restrict__ dtb,
                                                   float* __restrict__ gk,
                                                   float* __restrict__ beta) {
  __shared__ float row[DD];
  const long r = blockIdx.x;   // b*L + l
  for (int i = threadIdx.x; i < DD; i += 256) row[i] = hs[r * DD + i];
  __syncthreads();
  if (threadIdx.x < 2 * HH) {
    const int h = threadIdx.x % HH;
    const bool isb = threadIdx.x >= HH;
    const float* w = (isb ? Wb : Wgk) + (long)h * DD;
    float acc = 0.f;
    for (int i = 0; i < DD; ++i) acc += row[i] * w[i];
    if (isb) {
      beta[r * HH + h] = 1.f / (1.f + __expf(-(acc + b_b[h])));
    } else {
      float x = acc + dtb[h];
      float sp = (x > 20.f) ? x : log1pf(__expf(x));
      gk[r * HH + h] = -__expf(A_log[h]) * sp;
    }
  }
}

// ---------------- per-head L2 norm (rows of 128, wave32) ----------------
__global__ __launch_bounds__(256) void l2norm_kernel(float* __restrict__ x, long nrows) {
  long rowi = (long)blockIdx.x * 8 + (threadIdx.x >> 5);
  int lane = threadIdx.x & 31;
  if (rowi >= nrows) return;
  float* p = x + rowi * HKD;
  float s = 0.f;
  for (int i = lane; i < HKD; i += 32) { float v = p[i]; s += v * v; }
#pragma unroll
  for (int off = 16; off > 0; off >>= 1) s += __shfl_xor(s, off, 32);
  float inv = rsqrtf(fmaxf(s, 1e-24f));   // == 1/clip(norm,1e-12)
  for (int i = lane; i < HKD; i += 32) p[i] *= inv;
}

// ---------------- chunked gated delta rule scan ----------------
// one workgroup per (b,h); S[128][256] fp32 lives in LDS across all 32 chunks
__global__ __launch_bounds__(256) void scan_kernel(const float* __restrict__ Q,
                                                   const float* __restrict__ Kx,
                                                   const float* __restrict__ V,
                                                   const float* __restrict__ GK,
                                                   const float* __restrict__ BETA,
                                                   float* __restrict__ O) {
  const int b = blockIdx.x / HH, h = blockIdx.x % HH;
  const int tid = threadIdx.x, lane = tid & 31, wid = tid >> 5;
  const int rhi = (lane >> 4) * 8, cnl = lane & 15;

  extern __shared__ char smem[];
  float*  S   = (float*)smem;                  // [128][256]
  bf16_t* qb  = (bf16_t*)(S + HKD * HVD);      // [64][128]
  bf16_t* kb  = qb + CKC * HKD;                // [64][128]
  bf16_t* vb  = kb + CKC * HKD;                // [64][256]
  float*  U   = (float*)(vb + CKC * HVD);      // [64][256]
  float*  Am  = U + CKC * HVD;                 // [64][64]
  float*  gam = Am + CKC * CKC;                // [64]
  float*  eg  = gam + CKC;                     // [64]
  float*  bet = eg + CKC;                      // [64]

  for (int i = tid; i < HKD * HVD; i += 256) S[i] = 0.f;
  __syncthreads();

  for (int c = 0; c < NCH; ++c) {
    const long rowb = (long)b * LLEN + (long)c * CKC;
    // (a) load raw gk/beta
    if (tid < CKC) {
      gam[tid] = GK[(rowb + tid) * HH + h];
      bet[tid] = BETA[(rowb + tid) * HH + h];
    }
    // (b) stage q,k,v -> bf16 LDS
    for (int i = tid; i < CKC * HKD; i += 256) {
      int r = i >> 7, d = i & 127;
      long gi = (rowb + r) * DKK + (long)h * HKD + d;
      qb[i] = (bf16_t)Q[gi];
      kb[i] = (bf16_t)Kx[gi];
    }
    for (int i = tid; i < CKC * HVD; i += 256) {
      int r = i >> 8, d = i & 255;
      vb[i] = (bf16_t)V[(rowb + r) * DVV + (long)h * HVD + d];
    }
    __syncthreads();
    if (tid == 0) {   // cumsum -> gamma
      float s = 0.f;
      for (int i = 0; i < CKC; ++i) { s += gam[i]; gam[i] = s; }
    }
    __syncthreads();
    if (tid < CKC) eg[tid] = __expf(gam[tid]);
    __syncthreads();

    // (c) A = tril(beta_i * exp(g_i-g_j) * k k^T, -1)
    for (int ti = wid; ti < 16; ti += 8) {
      int tm = ti >> 2, tn = ti & 3;
      v8f acc = {};
      for (int kk = 0; kk < HKD; kk += 32) {
        bf16x16 a  = frag_a_lds(kb, HKD, tm * 16, kk, lane);
        bf16x16 bm = frag_b_lds(kb, HKD, tn * 16, kk, lane);   // k^T
        acc = wmma_bf16(a, bm, acc);
      }
#pragma unroll
      for (int r = 0; r < 8; ++r) {
        int i = tm * 16 + rhi + r, j = tn * 16 + cnl;
        Am[i * CKC + j] = (j < i) ? bet[i] * __expf(gam[i] - gam[j]) * acc[r] : 0.f;
      }
    }
    __syncthreads();

    // (d) rhs: U = beta*v - beta*eg*(k @ S)
    for (int ti = wid; ti < 64; ti += 8) {
      int tm = ti >> 4, tn = ti & 15;
      v8f acc = {};
      for (int kk = 0; kk < HKD; kk += 32) {
        bf16x16 a  = frag_a_lds(kb, HKD, tm * 16, kk, lane);
        bf16x16 bm = frag_b_f32(S, HVD, 1, kk, tn * 16, lane);
        acc = wmma_bf16(a, bm, acc);
      }
#pragma unroll
      for (int r = 0; r < 8; ++r) {
        int i = tm * 16 + rhi + r, n = tn * 16 + cnl;
        U[i * HVD + n] = bet[i] * (float)vb[i * HVD + n] - bet[i] * eg[i] * acc[r];
      }
    }
    __syncthreads();

    // (e) forward substitution (I+A) U = rhs ; columns independent (1 col/thread)
    {
      const int col = tid;
      for (int i = 1; i < CKC; ++i) {
        float u = U[i * HVD + col];
        for (int j = 0; j < i; ++j) u -= Am[i * CKC + j] * U[j * HVD + col];
        U[i * HVD + col] = u;
      }
    }
    __syncthreads();

    // (f) attn = tril(exp(g_i-g_j) * q k^T) -> reuse Am
    for (int ti = wid; ti < 16; ti += 8) {
      int tm = ti >> 2, tn = ti & 3;
      v8f acc = {};
      for (int kk = 0; kk < HKD; kk += 32) {
        bf16x16 a  = frag_a_lds(qb, HKD, tm * 16, kk, lane);
        bf16x16 bm = frag_b_lds(kb, HKD, tn * 16, kk, lane);
        acc = wmma_bf16(a, bm, acc);
      }
#pragma unroll
      for (int r = 0; r < 8; ++r) {
        int i = tm * 16 + rhi + r, j = tn * 16 + cnl;
        Am[i * CKC + j] = (j <= i) ? __expf(gam[i] - gam[j]) * acc[r] : 0.f;
      }
    }
    __syncthreads();

    // (g) o = eg * (q @ S) + attn @ U
    for (int ti = wid; ti < 64; ti += 8) {
      int tm = ti >> 4, tn = ti & 15;
      v8f acc = {};
      for (int kk = 0; kk < HKD; kk += 32) {
        bf16x16 a  = frag_a_lds(qb, HKD, tm * 16, kk, lane);
        bf16x16 bm = frag_b_f32(S, HVD, 1, kk, tn * 16, lane);
        acc = wmma_bf16(a, bm, acc);
      }
#pragma unroll
      for (int r = 0; r < 8; ++r) acc[r] *= eg[tm * 16 + rhi + r];
      for (int kk = 0; kk < CKC; kk += 32) {
        bf16x16 a  = frag_a_f32(Am, CKC, 1, tm * 16, kk, lane);
        bf16x16 bm = frag_b_f32(U, HVD, 1, kk, tn * 16, lane);
        acc = wmma_bf16(a, bm, acc);
      }
#pragma unroll
      for (int r = 0; r < 8; ++r) {
        int i = tm * 16 + rhi + r, n = tn * 16 + cnl;
        O[(rowb + i) * DVV + (long)h * HVD + n] = acc[r];
      }
    }
    __syncthreads();

    // (h) kg = k * exp(g_last - gamma) (overwrite kb)
    for (int i = tid; i < CKC * HKD; i += 256) {
      int r = i >> 7;
      kb[i] = (bf16_t)((float)kb[i] * __expf(gam[CKC - 1] - gam[r]));
    }
    __syncthreads();

    // (i) S = exp(g_last)*S + kg^T @ U
    const float egl = __expf(gam[CKC - 1]);
    for (int ti = wid; ti < 128; ti += 8) {
      int tm = ti >> 4, tn = ti & 15;
      v8f acc = {};
      for (int kk = 0; kk < CKC; kk += 32) {
        bf16x16 a  = frag_a_bf(kb, 1, HKD, tm * 16, kk, lane);   // kg transposed
        bf16x16 bm = frag_b_f32(U, HVD, 1, kk, tn * 16, lane);
        acc = wmma_bf16(a, bm, acc);
      }
#pragma unroll
      for (int r = 0; r < 8; ++r) {
        int m = tm * 16 + rhi + r, n = tn * 16 + cnl;
        S[m * HVD + n] = egl * S[m * HVD + n] + acc[r];
      }
    }
    __syncthreads();
  }
}

// ---------------- LayerNorm -> SiLU -> * g, output bf16 for final GEMM ----------------
__global__ __launch_bounds__(256) void post_kernel(const float* __restrict__ Ov,
                                                   const float* __restrict__ G,
                                                   const float* __restrict__ lng,
                                                   const float* __restrict__ lnb,
                                                   bf16_t* __restrict__ og) {
  __shared__ float s1[256], s2[256];
  const long row = blockIdx.x;    // (b*L + l)*H + h
  const int t = threadIdx.x;
  float v = Ov[row * HVD + t];
  s1[t] = v; s2[t] = v * v;
  __syncthreads();
  for (int off = 128; off > 0; off >>= 1) {
    if (t < off) { s1[t] += s1[t + off]; s2[t] += s2[t + off]; }
    __syncthreads();
  }
  float mu = s1[0] * (1.f / HVD);
  float var = s2[0] * (1.f / HVD) - mu * mu;
  float xh = (v - mu) * rsqrtf(var + 1e-5f) * lng[t] + lnb[t];
  float sil = xh / (1.f + __expf(-xh));
  og[row * HVD + t] = (bf16_t)(sil * G[row * HVD + t]);
}

// ---------------- host ----------------
extern "C" void kernel_launch(void* const* d_in, const int* in_sizes, int n_in,
                              void* d_out, int out_size, void* d_ws, size_t ws_size,
                              hipStream_t stream) {
  (void)in_sizes; (void)n_in; (void)out_size; (void)ws_size;
  const float* hs    = (const float*)d_in[0];
  const float* Wq    = (const float*)d_in[1];
  const float* Wk    = (const float*)d_in[2];
  const float* Wv    = (const float*)d_in[3];
  const float* Wg    = (const float*)d_in[4];
  const float* Wo    = (const float*)d_in[5];
  const float* cqw   = (const float*)d_in[6];
  const float* ckw   = (const float*)d_in[7];
  const float* cvw   = (const float*)d_in[8];
  const float* Wgk   = (const float*)d_in[9];
  const float* Wbm   = (const float*)d_in[10];
  const float* b_b   = (const float*)d_in[11];
  const float* A_log = (const float*)d_in[12];
  const float* dtb   = (const float*)d_in[13];
  const float* lng   = (const float*)d_in[14];
  const float* lnb   = (const float*)d_in[15];
  float* out = (float*)d_out;

  char* p = (char*)d_ws;
  auto take = [&](size_t bytes) -> char* {
    char* r = p; p += (bytes + 255) & ~(size_t)255; return r;
  };
  const size_t ML = (size_t)BB * LLEN;   // 4096 rows

  bf16_t* hsb  = (bf16_t*)take(ML * DD * 2);
  bf16_t* wqb  = (bf16_t*)take((size_t)DKK * DD * 2);
  bf16_t* wkb  = (bf16_t*)take((size_t)DKK * DD * 2);
  bf16_t* wvb  = (bf16_t*)take((size_t)DVV * DD * 2);
  bf16_t* wgb  = (bf16_t*)take((size_t)DVV * DD * 2);
  bf16_t* wob  = (bf16_t*)take((size_t)DD * DVV * 2);
  bf16_t* cqb  = (bf16_t*)take((size_t)4 * DKK * DKK * 2);
  bf16_t* ckb  = (bf16_t*)take((size_t)4 * DKK * DKK * 2);
  bf16_t* cvb  = (bf16_t*)take((size_t)4 * DVV * DVV * 2);
  bf16_t* qpre = (bf16_t*)take(ML * DKK * 2);   // sizes are 256B multiples ->
  bf16_t* kpre = (bf16_t*)take(ML * DKK * 2);   // qpre/kpre/vpre contiguous
  bf16_t* vpre = (bf16_t*)take(ML * DVV * 2);
  float*  qf   = (float*)take(ML * DKK * 4);
  float*  kf   = (float*)take(ML * DKK * 4);
  float*  vf   = (float*)take(ML * DVV * 4);
  float*  gkb  = (float*)take(ML * HH * 4);
  float*  btb  = (float*)take(ML * HH * 4);
  // reuse of dead regions (stream-ordered so safe):
  float*  gf  = (float*)qpre;   // g proj (after convs consume q/k/v pre-acts)
  float*  of  = (float*)cvb;    // scan output (after conv_v consumes weights)
  bf16_t* ogb = (bf16_t*)cqb;   // gated output (after conv_q/k consume weights)

  dim3 blk(256);
  // 1) casts + conv weight packing
  cast_f32_to_bf16<<<2048, blk, 0, stream>>>(hs, hsb, (long)ML * DD);
  cast_f32_to_bf16<<<1024, blk, 0, stream>>>(Wq, wqb, (long)DKK * DD);
  cast_f32_to_bf16<<<1024, blk, 0, stream>>>(Wk, wkb, (long)DKK * DD);
  cast_f32_to_bf16<<<2048, blk, 0, stream>>>(Wv, wvb, (long)DVV * DD);
  cast_f32_to_bf16<<<2048, blk, 0, stream>>>(Wg, wgb, (long)DVV * DD);
  cast_f32_to_bf16<<<2048, blk, 0, stream>>>(Wo, wob, (long)DD * DVV);
  pack_conv_w<<<2048, blk, 0, stream>>>(cqw, cqb, DKK);
  pack_conv_w<<<2048, blk, 0, stream>>>(ckw, ckb, DKK);
  pack_conv_w<<<4096, blk, 0, stream>>>(cvw, cvb, DVV);
  // 2) q/k/v projections (bf16 out -> conv input)
  gemm_nt<bf16_t><<<dim3(DKK / 128, ML / 128), blk, 0, stream>>>(hsb, wqb, qpre, (int)ML, DKK, DD);
  gemm_nt<bf16_t><<<dim3(DKK / 128, ML / 128), blk, 0, stream>>>(hsb, wkb, kpre, (int)ML, DKK, DD);
  gemm_nt<bf16_t><<<dim3(DVV / 128, ML / 128), blk, 0, stream>>>(hsb, wvb, vpre, (int)ML, DVV, DD);
  // 3) causal channel-mixing convolutions
  conv_gemm<<<dim3(DKK / 128, ML / 128), blk, 0, stream>>>(qpre, cqb, qf, DKK);
  conv_gemm<<<dim3(DKK / 128, ML / 128), blk, 0, stream>>>(kpre, ckb, kf, DKK);
  conv_gemm<<<dim3(DVV / 128, ML / 128), blk, 0, stream>>>(vpre, cvb, vf, DVV);
  // 4) gates + per-head l2norm
  gate_kernel<<<(unsigned)ML, blk, 0, stream>>>(hs, Wgk, Wbm, b_b, A_log, dtb, gkb, btb);
  l2norm_kernel<<<(unsigned)((ML * HH) / 8), blk, 0, stream>>>(qf, (long)ML * HH);
  l2norm_kernel<<<(unsigned)((ML * HH) / 8), blk, 0, stream>>>(kf, (long)ML * HH);
  // 5) g projection (reuses pre-act storage)
  gemm_nt<float><<<dim3(DVV / 128, ML / 128), blk, 0, stream>>>(hsb, wgb, gf, (int)ML, DVV, DD);
  // 6) chunked gated delta-rule scan (dynamic LDS: S + tiles, ~273 KB < 320 KB WGP LDS)
  size_t smem = (size_t)HKD * HVD * 4 + (size_t)2 * CKC * HKD * 2 + (size_t)CKC * HVD * 2 +
                (size_t)CKC * HVD * 4 + (size_t)CKC * CKC * 4 + 3 * CKC * 4;
  scan_kernel<<<BB * HH, blk, smem, stream>>>(qf, kf, vf, gkb, btb, of);
  // 7) LayerNorm/SiLU/gate
  post_kernel<<<(unsigned)(ML * HH), blk, 0, stream>>>(of, gf, lng, lnb, ogb);
  // 8) output projection -> d_out (fp32)
  gemm_nt<float><<<dim3(DD / 128, ML / 128), blk, 0, stream>>>(ogb, wob, out, (int)ML, DD, DVV);
}
